// GCN_81973745811568
// MI455X (gfx1250) — matrix operand order repaired and checked by reading, not compiled, not run on hardware
//
#include <hip/hip_runtime.h>
#include <hip/hip_bf16.h>

// ---------- types ----------
typedef __bf16 bf16_t;
typedef __attribute__((ext_vector_type(16))) __bf16 v16bf;
typedef __attribute__((ext_vector_type(8)))  __bf16 v8bf;
typedef __attribute__((ext_vector_type(8)))  float  v8f;

// ---------- problem constants ----------
#define NN   50000
#define EE   800000
#define FIN  512
#define HID  512
#define CC   256

// =====================================================================
// small utility kernels
// =====================================================================
__global__ void k_zero_f32(float* __restrict__ p, size_t n) {
    size_t i = (size_t)blockIdx.x * blockDim.x + threadIdx.x;
    size_t stride = (size_t)gridDim.x * blockDim.x;
    for (; i < n; i += stride) p[i] = 0.0f;
}

__global__ void k_deg_init(int* __restrict__ deg, int n) {
    int i = blockIdx.x * blockDim.x + threadIdx.x;
    if (i < n) deg[i] = 0;
}

__global__ void k_deg_count(const int* __restrict__ dst, int* __restrict__ deg, int e) {
    int i = blockIdx.x * blockDim.x + threadIdx.x;
    if (i < e) atomicAdd(&deg[dst[i]], 1);
}

__global__ void k_dinv(const int* __restrict__ deg, float* __restrict__ dinv, int n) {
    int i = blockIdx.x * blockDim.x + threadIdx.x;
    if (i < n) dinv[i] = rsqrtf((float)deg[i] + 1.0f);
}

__global__ void k_cvt_x(const float* __restrict__ x, bf16_t* __restrict__ xb, size_t n) {
    size_t i = (size_t)blockIdx.x * blockDim.x + threadIdx.x;
    size_t stride = (size_t)gridDim.x * blockDim.x;
    for (; i < n; i += stride) xb[i] = (bf16_t)x[i];
}

// Wt[o*K + k] = bf16(W[k*OUTD + o])   (column-major bf16 weights)
__global__ void k_cvt_w_t(const float* __restrict__ W, bf16_t* __restrict__ Wt,
                          int K, int OUTD) {
    size_t i = (size_t)blockIdx.x * blockDim.x + threadIdx.x;
    size_t total = (size_t)K * OUTD;
    size_t stride = (size_t)gridDim.x * blockDim.x;
    for (; i < total; i += stride) {
        int o = (int)(i / K);
        int k = (int)(i % K);
        Wt[i] = (bf16_t)W[(size_t)k * OUTD + o];
    }
}

// =====================================================================
// WMMA bf16 GEMM:  H[M, outDim] = Xb[M, 512] * W (Wt is col-major bf16)
// Block = 256 threads = 8 waves. Wave w -> row tile (blockIdx.x*8 + w),
// covering 16 rows x 64 cols (4 accumulators). K = 512 fixed.
// =====================================================================
__global__ void __launch_bounds__(256)
k_gemm_bf16(const bf16_t* __restrict__ Xb, const bf16_t* __restrict__ Wt,
            float* __restrict__ H, int nRows, int outDim) {
    const int wave = threadIdx.x >> 5;
    const int lane = threadIdx.x & 31;
    const int l15  = lane & 15;
    const int khalf = lane >> 4;           // 0 or 1

    const int rowTile = blockIdx.x * 8 + wave;
    const int rowBase = rowTile * 16;
    if (rowBase >= nRows) return;          // wave-uniform exit, EXEC stays full

    const int colBase = blockIdx.y * 64;

    v8f acc0 = {}, acc1 = {}, acc2 = {}, acc3 = {};

    const bf16_t* __restrict__ arow = Xb + (size_t)(rowBase + l15) * 512;
    const bf16_t* __restrict__ b0p = Wt + (size_t)(colBase +  0 + l15) * 512 + khalf * 16;
    const bf16_t* __restrict__ b1p = Wt + (size_t)(colBase + 16 + l15) * 512 + khalf * 16;
    const bf16_t* __restrict__ b2p = Wt + (size_t)(colBase + 32 + l15) * 512 + khalf * 16;
    const bf16_t* __restrict__ b3p = Wt + (size_t)(colBase + 48 + l15) * 512 + khalf * 16;

    #pragma unroll 4
    for (int kb = 0; kb < 512; kb += 32) {
        // A fragment: lanes 0-15 hold K = kb+{0..15}; lanes 16-31 K = kb+{16..31}
        v16bf a = *(const v16bf*)(arow + kb + khalf * 16);

        // B fragments: 16 contiguous K values per lane (col-major weights)
        v16bf b0 = *(const v16bf*)(b0p + kb);
        v16bf b1 = *(const v16bf*)(b1p + kb);
        v16bf b2 = *(const v16bf*)(b2p + kb);
        v16bf b3 = *(const v16bf*)(b3p + kb);

        acc0 = __builtin_amdgcn_wmma_f32_16x16x32_bf16(false, a, false, b0, (short)0, acc0, false, false);
        acc1 = __builtin_amdgcn_wmma_f32_16x16x32_bf16(false, a, false, b1, (short)0, acc1, false, false);
        acc2 = __builtin_amdgcn_wmma_f32_16x16x32_bf16(false, a, false, b2, (short)0, acc2, false, false);
        acc3 = __builtin_amdgcn_wmma_f32_16x16x32_bf16(false, a, false, b3, (short)0, acc3, false, false);
    }

    // C/D layout: lane col = lane&15; elem j -> row = j + 8*khalf
    #pragma unroll
    for (int j = 0; j < 8; ++j) {
        int r = rowBase + j + 8 * khalf;
        float* __restrict__ hrow = H + (size_t)r * outDim + colBase + l15;
        hrow[ 0] = acc0[j];
        hrow[16] = acc1[j];
        hrow[32] = acc2[j];
        hrow[48] = acc3[j];
    }
}

// =====================================================================
// Edge scatter: AGG[dst, :] += H[src, :] * dinv[src]*dinv[dst]
// One block per edge; threads stride the feature dim.
// =====================================================================
__global__ void __launch_bounds__(256)
k_scatter(const int* __restrict__ src, const int* __restrict__ dst,
          const float* __restrict__ dinv, const float* __restrict__ H,
          float* __restrict__ AGG, int F) {
    int e = blockIdx.x;
    int s = src[e];
    int d = dst[e];
    float coef = dinv[s] * dinv[d];
    const float* __restrict__ hs = H + (size_t)s * F;
    float* __restrict__ ad = AGG + (size_t)d * F;
    for (int f = threadIdx.x; f < F; f += blockDim.x) {
        unsafeAtomicAdd(&ad[f], hs[f] * coef);   // global_atomic_add_f32
    }
}

// =====================================================================
// Epilogue (layers 0,1): out = swish(AGG + H*dinv^2 + b)  -> bf16 (next GEMM input)
// =====================================================================
__global__ void k_fin_swish(const float* __restrict__ AGG, const float* __restrict__ H,
                            const float* __restrict__ dinv, const float* __restrict__ b,
                            bf16_t* __restrict__ out, int fshift, int fmask, size_t total) {
    size_t i = (size_t)blockIdx.x * blockDim.x + threadIdx.x;
    size_t stride = (size_t)gridDim.x * blockDim.x;
    for (; i < total; i += stride) {
        int node = (int)(i >> fshift);
        int f    = (int)(i & fmask);
        float dv = dinv[node];
        float v  = AGG[i] + H[i] * dv * dv + b[f];
        float sw = v / (1.0f + expf(-v));        // v * sigmoid(v)
        out[i] = (bf16_t)sw;
    }
}

// =====================================================================
// Epilogue (layer 2): v = AGG + H*dinv^2 + b2, then log_softmax over C=256.
// One block (256 threads) per node.
// =====================================================================
__global__ void __launch_bounds__(256)
k_fin_lsm(const float* __restrict__ AGG, const float* __restrict__ H,
          const float* __restrict__ dinv, const float* __restrict__ b,
          float* __restrict__ out) {
    __shared__ float sm[256];
    int node = blockIdx.x;
    int f = threadIdx.x;
    size_t idx = (size_t)node * CC + f;
    float dv = dinv[node];
    float v = AGG[idx] + H[idx] * dv * dv + b[f];

    // max reduction
    sm[f] = v;
    __syncthreads();
    for (int off = 128; off > 0; off >>= 1) {
        if (f < off) sm[f] = fmaxf(sm[f], sm[f + off]);
        __syncthreads();
    }
    float m = sm[0];
    __syncthreads();

    // sum(exp) reduction
    sm[f] = expf(v - m);
    __syncthreads();
    for (int off = 128; off > 0; off >>= 1) {
        if (f < off) sm[f] += sm[f + off];
        __syncthreads();
    }
    float lse = m + logf(sm[0]);

    out[idx] = v - lse;
}

// =====================================================================
// host-side launch
// =====================================================================
extern "C" void kernel_launch(void* const* d_in, const int* in_sizes, int n_in,
                              void* d_out, int out_size, void* d_ws, size_t ws_size,
                              hipStream_t stream) {
    const float* x   = (const float*)d_in[0];
    const int*   ei  = (const int*)d_in[1];        // [2, E]
    const float* W0  = (const float*)d_in[2];
    const float* b0  = (const float*)d_in[3];
    const float* W1  = (const float*)d_in[4];
    const float* b1  = (const float*)d_in[5];
    const float* W2  = (const float*)d_in[6];
    const float* b2  = (const float*)d_in[7];
    float* out = (float*)d_out;

    const int* src = ei;
    const int* dst = ei + EE;

    // ---- carve workspace (256B aligned) ----
    size_t off = 0;
    char* ws = (char*)d_ws;
    auto carve = [&](size_t bytes) -> void* {
        void* p = ws + off;
        off += (bytes + 255) & ~(size_t)255;
        return p;
    };
    float*  dinv = (float*)carve((size_t)NN * 4);
    int*    degi = (int*)  carve((size_t)NN * 4);
    bf16_t* Xb   = (bf16_t*)carve((size_t)NN * HID * 2);
    float*  H    = (float*) carve((size_t)NN * HID * 4);
    float*  AGG  = (float*) carve((size_t)NN * HID * 4);
    bf16_t* W0t  = (bf16_t*)carve((size_t)FIN * HID * 2);
    bf16_t* W1t  = (bf16_t*)carve((size_t)HID * HID * 2);
    bf16_t* W2t  = (bf16_t*)carve((size_t)HID * CC * 2);
    (void)ws_size; (void)n_in; (void)in_sizes; (void)out_size;

    const int T = 256;

    // ---- degrees -> dinv ----
    k_deg_init <<<(NN + T - 1) / T, T, 0, stream>>>(degi, NN);
    k_deg_count<<<(EE + T - 1) / T, T, 0, stream>>>(dst, degi, EE);
    k_dinv     <<<(NN + T - 1) / T, T, 0, stream>>>(degi, dinv, NN);

    // ---- weights -> bf16, transposed to col-major ----
    k_cvt_w_t<<<2048, T, 0, stream>>>(W0, W0t, FIN, HID);
    k_cvt_w_t<<<2048, T, 0, stream>>>(W1, W1t, HID, HID);
    k_cvt_w_t<<<1024, T, 0, stream>>>(W2, W2t, HID, CC);

    // ---- x -> bf16 ----
    k_cvt_x<<<4096, T, 0, stream>>>(x, Xb, (size_t)NN * FIN);

    const int rowTiles  = (NN + 15) / 16;           // 3125
    const int gemmGridX = (rowTiles + 7) / 8;       // 391

    // =================== layer 0 ===================
    k_gemm_bf16<<<dim3(gemmGridX, HID / 64), T, 0, stream>>>(Xb, W0t, H, NN, HID);
    k_zero_f32 <<<4096, T, 0, stream>>>(AGG, (size_t)NN * HID);
    k_scatter  <<<EE, T, 0, stream>>>(src, dst, dinv, H, AGG, HID);
    k_fin_swish<<<4096, T, 0, stream>>>(AGG, H, dinv, b0, Xb, 9, 511, (size_t)NN * HID);

    // =================== layer 1 ===================
    k_gemm_bf16<<<dim3(gemmGridX, HID / 64), T, 0, stream>>>(Xb, W1t, H, NN, HID);
    k_zero_f32 <<<4096, T, 0, stream>>>(AGG, (size_t)NN * HID);
    k_scatter  <<<EE, T, 0, stream>>>(src, dst, dinv, H, AGG, HID);
    k_fin_swish<<<4096, T, 0, stream>>>(AGG, H, dinv, b1, Xb, 9, 511, (size_t)NN * HID);

    // =================== layer 2 + log_softmax ===================
    k_gemm_bf16<<<dim3(gemmGridX, CC / 64), T, 0, stream>>>(Xb, W2t, H, NN, CC);
    k_zero_f32 <<<4096, T, 0, stream>>>(AGG, (size_t)NN * CC);
    k_scatter  <<<EE, T, 0, stream>>>(src, dst, dinv, H, AGG, CC);
    k_fin_lsm  <<<NN, CC, 0, stream>>>(AGG, H, dinv, b2, out);
}
